// ChamferDistance_75685913690395
// MI455X (gfx1250) — matrix operand order repaired and checked by reading, not compiled, not run on hardware
//
#include <hip/hip_runtime.h>

// ---------------------------------------------------------------------------
// Chamfer distance on MI455X (gfx1250, wave32).
//   out[n] = 0.5 * ( mean_p min_q ||x_np - y_nq|| + mean_q min_p ||x_np - y_nq|| )
// Strategy: distance-GEMM on v_wmma_f32_16x16x32_bf16 with bf16 hi/lo split
// (fp32-grade accuracy, 3 WMMAs per K-chunk), streaming min over 16-wide
// column tiles, norms in exact fp32. Fully deterministic (no FP atomics).
// ---------------------------------------------------------------------------

typedef float  v8f    __attribute__((ext_vector_type(8)));
typedef __bf16 v16bf  __attribute__((ext_vector_type(16)));
typedef __bf16 bf16x4 __attribute__((ext_vector_type(4)));

static __device__ __forceinline__ v8f wmma_bf16(v16bf a, v16bf b, v8f c) {
  // (neg_a, A, neg_b, B, c_mod, C, reuse_a, reuse_b)
  return __builtin_amdgcn_wmma_f32_16x16x32_bf16(false, a, false, b,
                                                 (short)0, c, false, false);
}

__global__ __launch_bounds__(256)
void ChamferDistance_dir_kernel(const float* __restrict__ X,
                                const float* __restrict__ Y,
                                float* __restrict__ partials)
{
  constexpr int P = 4096;
  constexpr int D = 128;
  constexpr int PANEL = 64;           // B points staged per iteration
  constexpr int NPANEL = P / PANEL;   // 64

  const int tileIdx = blockIdx.x;     // 0..31 : 128-row slice of A
  const int batch   = blockIdx.y;     // 0..7
  const int dir     = blockIdx.z;     // 0: A=X,B=Y   1: A=Y,B=X

  const float* __restrict__ Aptr = dir ? Y : X;
  const float* __restrict__ Bptr = dir ? X : Y;

  // B panel staged in WMMA B-fragment order: per (qt,c) tile-chunk, 32 lanes
  // each own 16 contiguous bf16 (32B) -> conflict-light ds_load_b128 pairs.
  __shared__ __align__(32) __bf16 ldsHi[PANEL * D];   // 16 KB
  __shared__ __align__(32) __bf16 ldsLo[PANEL * D];   // 16 KB
  __shared__ float y2s[PANEL];
  __shared__ float waveSums[8];

  const int tid  = threadIdx.x;
  const int lane = tid & 31;
  const int wave = tid >> 5;

  // ---- Build this wave's A fragments (16 rows x 128 dims), bf16 hi/lo. ----
  // ISA 16-bit A 16x32 layout: lane m (0-15) row m, K = base..base+7 in V0-3
  // and base+16..base+23 in V4-7, base = (lane>=16)*8.
  const int m     = lane & 15;
  const int kbase = (lane >> 4) * 8;
  const float* arow = Aptr + (size_t)batch * P * D
                    + (size_t)(tileIdx * 128 + wave * 16 + m) * D;
  v16bf Ahi[4], Alo[4];
  float part = 0.f;                 // exact fp32 partial of ||row||^2
#pragma unroll
  for (int c = 0; c < 4; ++c) {
    const float* p0 = arow + c * 32 + kbase;
#pragma unroll
    for (int t = 0; t < 8; ++t) {
      float v0 = p0[t];
      float v1 = p0[16 + t];
      part += v0 * v0 + v1 * v1;
      __bf16 h0 = (__bf16)v0;
      __bf16 h1 = (__bf16)v1;
      Ahi[c][t]     = h0;
      Ahi[c][8 + t] = h1;
      Alo[c][t]     = (__bf16)(v0 - (float)h0);
      Alo[c][8 + t] = (__bf16)(v1 - (float)h1);
    }
  }
  // lanes l and l^16 jointly hold row (l&15): combine for full ||row||^2.
  float x2full = part + __shfl_xor(part, 16, 32);
  // Accumulator layout: lane holds column N=lane&15, rows v + 8*(lane>=16).
  const int hi8 = (lane >> 4) * 8;
  float x2v[8];
#pragma unroll
  for (int v = 0; v < 8; ++v) x2v[v] = __shfl(x2full, v + hi8, 32);

  float runMin[8];
#pragma unroll
  for (int v = 0; v < 8; ++v) runMin[v] = 3.0e38f;

  // Staging roles: 4 threads per B point (one 32-dim chunk each).
  const int q    = tid >> 2;     // 0..63 point within panel
  const int cc   = tid & 3;      // K chunk 0..3
  const int qt_s = q >> 4;
  const int n_s  = q & 15;
  const float* Bbatch = Bptr + (size_t)batch * P * D;

  for (int panel = 0; panel < NPANEL; ++panel) {
    // ---- stage: f32 -> bf16 hi/lo into fragment-order LDS + ||y||^2 ----
    const float4* src =
        (const float4*)(Bbatch + (size_t)(panel * PANEL + q) * D + cc * 32);
    if (panel + 1 < NPANEL)
      __builtin_prefetch(
          Bbatch + (size_t)((panel + 1) * PANEL + q) * D + cc * 32, 0, 1);
    float ps = 0.f;
#pragma unroll
    for (int i = 0; i < 8; ++i) {
      float4 f = src[i];
      ps += f.x * f.x + f.y * f.y + f.z * f.z + f.w * f.w;
      const int half = (i >> 1) & 1;    // lane half: K bit 3 within chunk
      const int j    = (i & 1) * 4 + (i >> 2) * 8;  // position within lane's 16 bf16 (table order)
      const int idx  = (((qt_s * 4 + cc) * 32) + n_s + 16 * half) * 16 + j;
      __bf16 h0 = (__bf16)f.x, h1 = (__bf16)f.y,
             h2 = (__bf16)f.z, h3 = (__bf16)f.w;
      bf16x4 hv = {h0, h1, h2, h3};
      bf16x4 lv = {(__bf16)(f.x - (float)h0), (__bf16)(f.y - (float)h1),
                   (__bf16)(f.z - (float)h2), (__bf16)(f.w - (float)h3)};
      *(bf16x4*)(ldsHi + idx) = hv;
      *(bf16x4*)(ldsLo + idx) = lv;
    }
    // 4 threads share a point: combine exact fp32 norm, one plain store.
    ps += __shfl_xor(ps, 1, 32);
    ps += __shfl_xor(ps, 2, 32);
    if (cc == 0) y2s[q] = ps;
    __syncthreads();

    // ---- compute: 4 tiles of 16 B-points vs this wave's 16 A-rows ----
#pragma unroll
    for (int qt = 0; qt < 4; ++qt) {
      v8f acc = {0.f, 0.f, 0.f, 0.f, 0.f, 0.f, 0.f, 0.f};
#pragma unroll
      for (int c = 0; c < 4; ++c) {
        const int fb = ((qt * 4 + c) * 32 + lane) * 16;
        v16bf Bh = *(const v16bf*)(ldsHi + fb);
        v16bf Bl = *(const v16bf*)(ldsLo + fb);
        acc = wmma_bf16(Ahi[c], Bh, acc);   // hi*hi
        acc = wmma_bf16(Ahi[c], Bl, acc);   // hi*lo
        acc = wmma_bf16(Alo[c], Bh, acc);   // lo*hi  (lo*lo ~2^-16, dropped)
      }
      const float y2n = y2s[qt * 16 + (lane & 15)];
#pragma unroll
      for (int v = 0; v < 8; ++v) {
        float sq = x2v[v] + y2n - 2.0f * acc[v];
        sq = fmaxf(sq, 0.f);
        runMin[v] = fminf(runMin[v], sqrtf(sq));
      }
    }
    __syncthreads();   // LDS reused next panel
  }

  // ---- min across the 16 columns held by each half-wave ----
#pragma unroll
  for (int off = 1; off < 16; off <<= 1) {
#pragma unroll
    for (int v = 0; v < 8; ++v)
      runMin[v] = fminf(runMin[v], __shfl_xor(runMin[v], off, 32));
  }
  float s = 0.f;
#pragma unroll
  for (int v = 0; v < 8; ++v) s += runMin[v];       // 8 rows of this half
  s += __shfl_xor(s, 16, 32);                       // all 16 rows of the wave
  if (lane == 0) waveSums[wave] = s;
  __syncthreads();
  if (tid == 0) {
    float tot = 0.f;
#pragma unroll
    for (int w = 0; w < 8; ++w) tot += waveSums[w]; // fixed order: determinism
    partials[((size_t)dir * 8 + batch) * 32 + tileIdx] = tot;
  }
}

__global__ void ChamferDistance_reduce_kernel(const float* __restrict__ partials,
                                              float* __restrict__ out)
{
  const int n = threadIdx.x;
  if (n < 8) {
    float s = 0.f;
    for (int d = 0; d < 2; ++d)
      for (int b = 0; b < 32; ++b)
        s += partials[((size_t)d * 8 + n) * 32 + b];
    // (sum_p min + sum_q min) / (2 * 4096)
    out[n] = s * (1.0f / 8192.0f);
  }
}

extern "C" void kernel_launch(void* const* d_in, const int* in_sizes, int n_in,
                              void* d_out, int out_size, void* d_ws, size_t ws_size,
                              hipStream_t stream) {
  (void)in_sizes; (void)n_in; (void)out_size; (void)ws_size;
  const float* x = (const float*)d_in[0];   // (8, 4096, 128) f32
  const float* y = (const float*)d_in[1];   // (8, 4096, 128) f32
  float* out = (float*)d_out;               // (8,) f32
  float* ws  = (float*)d_ws;                // 512 floats of block partials

  dim3 grid(32, 8, 2);                      // row-tiles x batches x directions
  ChamferDistance_dir_kernel<<<grid, 256, 0, stream>>>(x, y, ws);
  ChamferDistance_reduce_kernel<<<1, 32, 0, stream>>>(ws, out);
}
